// ScaleDotProductAttention_10299331576252
// MI455X (gfx1250) — matrix operand order repaired and checked by reading, not compiled, not run on hardware
//
#include <hip/hip_runtime.h>

// ---------------------------------------------------------------------------
// Fused attention fwd for gfx1250 (CDNA5, wave32), materializing attn.
//   q,k,v: [B=4, H=16, S=2048, D=64] fp32
//   d_out: out [B,H,S,D] fp32  ++  attn [B,H,S,S] fp32
// One WG (4 waves / 128 threads) per (head, 64-query block).
// Matrix math: v_wmma_f32_16x16x32_bf16.
// Softmax stats lane-local in pass 1; single cross-lane merge.
// Raw v_exp_f32 / v_rcp_f32 (flush-below-2^-126 is exact enough for softmax).
// ---------------------------------------------------------------------------

typedef __attribute__((ext_vector_type(16))) __bf16 v16bf;
typedef __attribute__((ext_vector_type(8)))  float  v8f;

#define S_LEN 2048
#define DHEAD 64
#define QLD   68   // padded LDS row stride (bf16 elems) for Q tile
#define PLD   36   // padded LDS row stride (bf16 elems) for P scratch

// 1/sqrt(64) * log2(e): softmax computed in base-2 domain
#define SCL 0.1803368801111244f

// raw hardware transcendentals (no subnormal-range fixup)
__device__ __forceinline__ float fast_exp2(float x) { return __builtin_amdgcn_exp2f(x); }
__device__ __forceinline__ float fast_rcp(float x)  { return __builtin_amdgcn_rcpf(x); }

__device__ __forceinline__ float rmax16(float v) {
  v = fmaxf(v, __shfl_xor(v, 1, 32));
  v = fmaxf(v, __shfl_xor(v, 2, 32));
  v = fmaxf(v, __shfl_xor(v, 4, 32));
  v = fmaxf(v, __shfl_xor(v, 8, 32));
  return v;
}
__device__ __forceinline__ float rsum16(float v) {
  v += __shfl_xor(v, 1, 32);
  v += __shfl_xor(v, 2, 32);
  v += __shfl_xor(v, 4, 32);
  v += __shfl_xor(v, 8, 32);
  return v;
}

__global__ __launch_bounds__(128)
void attn_fwd_wmma(const float* __restrict__ q, const float* __restrict__ k,
                   const float* __restrict__ v, float* __restrict__ out,
                   float* __restrict__ attn)
{
  __shared__ __bf16 sQ[64 * QLD];        // 64x64 Q tile, bf16
  __shared__ __bf16 sP[4 * 16 * PLD];    // per-wave 16x32 P transpose scratch

  const int head = blockIdx.y;           // 0..63 (B*H)
  const int qb   = blockIdx.x;           // 0..31 query blocks of 64
  const int tid  = threadIdx.x;
  const int lane = tid & 31;
  const int wave = tid >> 5;
  const int row  = lane & 15;            // doubles as B-matrix column index
  const int hi   = (lane >= 16);
  const int kb   = hi ? 16 : 0;          // B-matrix K-half select

  const float* qg = q + ((size_t)head * S_LEN + (size_t)qb * 64) * DHEAD;
  const float* kg = k + (size_t)head * S_LEN * DHEAD;
  const float* vg = v + (size_t)head * S_LEN * DHEAD;
  float* outg  = out  + ((size_t)head * S_LEN + (size_t)qb * 64) * DHEAD;
  float* attng = attn + ((size_t)head * S_LEN + (size_t)qb * 64) * (size_t)S_LEN;

  // ---- Stage Q tile (64x64 fp32 -> bf16) into LDS cooperatively ----
  for (int i = tid; i < 64 * 64; i += 128) {
    int r = i >> 6, c = i & 63;
    sQ[r * QLD + c] = (__bf16)qg[r * 64 + c];
  }
  __syncthreads();

  // ---- Build per-wave WMMA A-operands for Q (rows wave*16..+15) ----
  // A 16x32 in contiguous K order (B matches): lane<16 holds K {0..15}, lane>=16 {16..31}
  const __bf16* sQw = sQ + (size_t)(wave * 16) * QLD;
  v16bf aQ0, aQ1;
  #pragma unroll
  for (int e = 0; e < 16; ++e) {
    int K = (hi ? 16 : 0) + e;
    aQ0[e] = sQw[row * QLD + K];
    aQ1[e] = sQw[row * QLD + 32 + K];
  }

  // ---- Pass 1: lane-local online softmax stats over this lane's columns ----
  float m[8], l[8];
  #pragma unroll
  for (int j = 0; j < 8; ++j) { m[j] = -3.0e38f; l[j] = 0.0f; }

  #pragma unroll 1
  for (int kt = 0; kt < S_LEN / 16; ++kt) {
    const float* kr = kg + (size_t)(kt * 16 + row) * DHEAD;
    // keep next K tile warm in GL2 (K gets re-streamed in pass 2)
    if (kt + 1 < S_LEN / 16)
      __builtin_prefetch(kg + (size_t)((kt + 1) * 16 + row) * DHEAD, 0, 0);

    v16bf b0, b1;                        // B = K^T, WMMA-K dim = d
    #pragma unroll
    for (int e = 0; e < 16; ++e) {
      b0[e] = (__bf16)kr[kb + e];
      b1[e] = (__bf16)kr[32 + kb + e];
    }
    v8f s = {};
    s = __builtin_amdgcn_wmma_f32_16x16x32_bf16(false, aQ0, false, b0, (short)0, s, false, false);
    s = __builtin_amdgcn_wmma_f32_16x16x32_bf16(false, aQ1, false, b1, (short)0, s, false, false);
    #pragma unroll
    for (int j = 0; j < 8; ++j) {
      float t  = s[j] * SCL;                    // base-2 score
      float mn = fmaxf(m[j], t);
      l[j] = l[j] * fast_exp2(m[j] - mn) + fast_exp2(t - mn);
      m[j] = mn;
    }
  }

  // ---- Single cross-lane merge of (m, l) across the 16-lane row groups ----
  float rl[8];
  #pragma unroll
  for (int j = 0; j < 8; ++j) {
    float mg = rmax16(m[j]);
    float lg = rsum16(l[j] * fast_exp2(m[j] - mg));
    m[j]  = mg;
    rl[j] = fast_rcp(lg);
  }

  // ---- Pass 2: recompute scores, write attn, accumulate O = P @ V ----
  v8f o[4] = {v8f{}, v8f{}, v8f{}, v8f{}};
  __bf16* sPw = sP + (size_t)(wave * 16) * PLD;

  #pragma unroll 1
  for (int kc = 0; kc < S_LEN / 32; ++kc) {
    #pragma unroll
    for (int sub = 0; sub < 2; ++sub) {
      const int kt = kc * 2 + sub;
      const float* kr = kg + (size_t)(kt * 16 + row) * DHEAD;
      v16bf b0, b1;
      #pragma unroll
      for (int e = 0; e < 16; ++e) {
        b0[e] = (__bf16)kr[kb + e];
        b1[e] = (__bf16)kr[32 + kb + e];
      }
      v8f s = {};
      s = __builtin_amdgcn_wmma_f32_16x16x32_bf16(false, aQ0, false, b0, (short)0, s, false, false);
      s = __builtin_amdgcn_wmma_f32_16x16x32_bf16(false, aQ1, false, b1, (short)0, s, false, false);
      #pragma unroll
      for (int j = 0; j < 8; ++j) {
        int   M = hi ? (j + 8) : j;
        float p = fast_exp2(s[j] * SCL - m[j]) * rl[j];
        attng[(size_t)(wave * 16 + M) * S_LEN + kt * 16 + row] = p;   // materialize attn
        sPw[M * PLD + sub * 16 + row] = (__bf16)p;                    // C-layout -> LDS
      }
    }
    // intra-wave LDS store->load ordering (CDNA5 split counter)
    asm volatile("s_wait_dscnt 0x0" ::: "memory");

    // reload P in A-matrix layout (16x32, WMMA-K dim = keys within chunk)
    v16bf aP;
    #pragma unroll
    for (int e = 0; e < 16; ++e) {
      int K = (hi ? 16 : 0) + e;
      aP[e] = sPw[row * PLD + K];
    }

    // O[16x64] += P(16x32) x V(32x64): 4 N-chunks of 16
    const float* vbase = vg + (size_t)(kc * 32 + kb) * DHEAD;
    #pragma unroll
    for (int n = 0; n < 4; ++n) {
      v16bf bv;
      #pragma unroll
      for (int e = 0; e < 16; ++e)
        bv[e] = (__bf16)vbase[(size_t)e * DHEAD + n * 16 + row];
      o[n] = __builtin_amdgcn_wmma_f32_16x16x32_bf16(false, aP, false, bv, (short)0, o[n], false, false);
    }
  }

  // ---- Epilogue: write O (fp32) ----
  #pragma unroll
  for (int j = 0; j < 8; ++j) {
    int M = hi ? (j + 8) : j;
    float* orow = outg + (size_t)(wave * 16 + M) * DHEAD;
    orow[ 0 + row] = o[0][j];
    orow[16 + row] = o[1][j];
    orow[32 + row] = o[2][j];
    orow[48 + row] = o[3][j];
  }
}

extern "C" void kernel_launch(void* const* d_in, const int* in_sizes, int n_in,
                              void* d_out, int out_size, void* d_ws, size_t ws_size,
                              hipStream_t stream) {
  (void)in_sizes; (void)n_in; (void)d_ws; (void)ws_size; (void)out_size;
  const float* q = (const float*)d_in[0];
  const float* k = (const float*)d_in[1];
  const float* v = (const float*)d_in[2];
  float* out  = (float*)d_out;                                  // [4,16,2048,64]
  float* attn = out + (size_t)4 * 16 * 2048 * 64;               // [4,16,2048,2048]

  dim3 grid(S_LEN / 64, 4 * 16);   // 32 query blocks x 64 heads
  dim3 block(128);                 // 4 waves (wave32)
  attn_fwd_wmma<<<grid, block, 0, stream>>>(q, k, v, out, attn);
}
